// SchemaItemClassifier_76304388981384
// MI455X (gfx1250) — matrix operand order repaired and checked by reading, not compiled, not run on hardware
//
#include <hip/hip_runtime.h>
#include <hip/hip_bf16.h>

// ---------------- problem constants ----------------
constexpr int kB  = 8;
constexpr int kS  = 4096;
constexpr int kH  = 768;
constexpr int kT  = 32;
constexpr int kLT = 8;
constexpr int kC  = 128;
constexpr int kLC = 16;
constexpr int kK3 = 3 * kH;          // 2304 (im2col K)

constexpr int P_T  = kB * kT * kLT;  // 2048 table positions
constexpr int P_C  = kB * kC * kLC;  // 16384 column positions
constexpr int NS_T = kB * kT;        // 256 table spans
constexpr int NS_C = kB * kC;        // 1024 column spans

// LDS B-panel tiling
constexpr int KCH   = 192;           // K elements staged per chunk
constexpr int LROW  = 200;           // padded LDS row (bank-conflict avoidance)
constexpr int LPLN  = 64 * LROW;     // elements per plane in LDS

// ---------------- workspace layout (bytes) ----------------
constexpr size_t EMBT_ROWS = (size_t)P_T + 16;
constexpr size_t EMBC_ROWS = (size_t)P_C + 16;
constexpr size_t OFF_EMBT_HI = 0;
constexpr size_t OFF_EMBT_LO = OFF_EMBT_HI + EMBT_ROWS * kH * 2;
constexpr size_t OFF_EMBC_HI = OFF_EMBT_LO + EMBT_ROWS * kH * 2;
constexpr size_t OFF_EMBC_LO = OFF_EMBC_HI + EMBC_ROWS * kH * 2;
constexpr size_t OFF_WT_HI   = OFF_EMBC_LO + EMBC_ROWS * kH * 2;
constexpr size_t OFF_WT_LO   = OFF_WT_HI + (size_t)kH * kK3 * 2;
constexpr size_t OFF_WC_HI   = OFF_WT_LO + (size_t)kH * kK3 * 2;
constexpr size_t OFF_WC_LO   = OFF_WC_HI + (size_t)kH * kK3 * 2;
constexpr size_t OFF_POOL_T  = OFF_WC_LO + (size_t)kH * kK3 * 2;
constexpr size_t OFF_POOL_C  = OFF_POOL_T + (size_t)NS_T * kH * 4;
// total = OFF_POOL_C + NS_C*kH*4  (~75 MB)

// ---------------- types ----------------
typedef __attribute__((ext_vector_type(16))) __bf16    v16bf;
typedef __attribute__((ext_vector_type(8)))  float     v8f;
typedef __attribute__((ext_vector_type(4)))  unsigned  u32x4;
typedef __attribute__((ext_vector_type(4)))  int       v4i;

union AFrag { v16bf v; u32x4 u[2]; };

__device__ __forceinline__ unsigned short f2bf(float f) {
    unsigned u = __float_as_uint(f);
    unsigned r = u + 0x7FFFu + ((u >> 16) & 1u);   // round-to-nearest-even
    return (unsigned short)(r >> 16);
}
__device__ __forceinline__ float bf2f(unsigned short h) {
    return __uint_as_float(((unsigned)h) << 16);
}

__device__ __forceinline__ v8f wmma_bf16(v16bf a, v16bf b, v8f c) {
    return __builtin_amdgcn_wmma_f32_16x16x32_bf16(
        /*neg_a=*/false, a, /*neg_b=*/false, b,
        /*c_mod=*/(short)0, c, /*reuse_a=*/false, /*reuse_b=*/false);
}

// -------- async global->LDS copy (CDNA5 ASYNCcnt path) with safe fallback ----
#if defined(__has_builtin)
#if __has_builtin(__builtin_amdgcn_global_load_async_to_lds_b128)
#define SCHEMA_HAVE_ASYNC 1
#endif
#endif

#ifdef SCHEMA_HAVE_ASYNC
__device__ __forceinline__ void cp16_async(const unsigned short* src, unsigned short* dst) {
    // Builtin signature (from hipcc diagnostic): param0 = int4 addrspace(1)*  (global src)
    __builtin_amdgcn_global_load_async_to_lds_b128(
        (__attribute__((address_space(1))) v4i*)(src),
        (__attribute__((address_space(3))) v4i*)(dst), 0, 0);
}
__device__ __forceinline__ void cp_async_wait() {
#if __has_builtin(__builtin_amdgcn_s_wait_asynccnt)
    __builtin_amdgcn_s_wait_asynccnt(0);
#else
    asm volatile("s_wait_asynccnt 0x0" ::: "memory");
#endif
}
#else
__device__ __forceinline__ void cp16_async(const unsigned short* src, unsigned short* dst) {
    *(u32x4*)dst = *(const u32x4*)src;     // load + ds_store fallback
}
__device__ __forceinline__ void cp_async_wait() {}
#endif

// ---------------- kernel 1: gather + mask + bf16 hi/lo split ----------------
__global__ void schema_gather_split(const float* __restrict__ hidden,
                                    const int* __restrict__ ids,
                                    const int* __restrict__ lens,
                                    unsigned short* __restrict__ hiP,
                                    unsigned short* __restrict__ loP,
                                    int nPerB, int L, int totalRows) {
    const int chunks = kH / 4;                       // 192 float4 chunks per row
    int tid = blockIdx.x * blockDim.x + threadIdx.x;
    int total = (totalRows + 1) * chunks;
    if (tid >= total) return;
    int p = tid / chunks;
    int c = (tid - p * chunks) * 4;
    float4 v = make_float4(0.f, 0.f, 0.f, 0.f);
    if (p < totalRows) {
        int l  = p % L;
        int sn = p / L;                              // flat span index b*nPerB+n
        int len = lens[sn];
        if (l < len) {
            int b   = sn / nPerB;
            int tok = ids[p];
            v = *(const float4*)(hidden + ((size_t)b * kS + tok) * kH + c);
        }
    }
    unsigned short h0 = f2bf(v.x), h1 = f2bf(v.y), h2 = f2bf(v.z), h3 = f2bf(v.w);
    unsigned short l0 = f2bf(v.x - bf2f(h0)), l1 = f2bf(v.y - bf2f(h1));
    unsigned short l2 = f2bf(v.z - bf2f(h2)), l3 = f2bf(v.w - bf2f(h3));
    size_t o = (size_t)p * kH + c;
    *(uint2*)(hiP + o) = make_uint2((unsigned)h0 | ((unsigned)h1 << 16),
                                    (unsigned)h2 | ((unsigned)h3 << 16));
    *(uint2*)(loP + o) = make_uint2((unsigned)l0 | ((unsigned)l1 << 16),
                                    (unsigned)l2 | ((unsigned)l3 << 16));
}

// ---------------- kernel 2: weight reorder (O,I,3) -> Wt[o][k*H+i], bf16 hi/lo ----
__global__ void schema_prep_w(const float* __restrict__ convW,
                              unsigned short* __restrict__ wHi,
                              unsigned short* __restrict__ wLo) {
    int idx = blockIdx.x * blockDim.x + threadIdx.x;   // over kH*kK3, exact grid
    int o = idx / kK3;
    int K = idx - o * kK3;
    int k = K / kH;
    int i = K - k * kH;
    float w = convW[((size_t)o * kH + i) * 3 + k];
    unsigned short h = f2bf(w);
    wHi[idx] = h;
    wLo[idx] = f2bf(w - bf2f(h));
}

// ---------------- kernel 3: im2col GEMM (WMMA bf16 split, LDS-staged B) ----------
// Block: 8 waves. Wave: 2 M-tiles (32 positions) x 4 N-tiles (64 channels).
// B panel (64 cols x 192 K, hi+lo) staged in LDS per chunk, shared by all 8 waves.
template <int LSPAN>
__global__ __launch_bounds__(256)
void schema_conv_gemm_pool(const unsigned short* __restrict__ embHi,
                           const unsigned short* __restrict__ embLo,
                           const unsigned short* __restrict__ wHi,
                           const unsigned short* __restrict__ wLo,
                           const float* __restrict__ convB,
                           const int* __restrict__ lens,
                           float* __restrict__ pooled,
                           int padRow) {
    __shared__ unsigned short ldsB[2 * LPLN];          // 2 planes * 64 * 200 * 2B = 50 KB

    const int tid   = threadIdx.x;
    const int lane  = tid & 31;
    const int wave  = tid >> 5;
    const int m     = lane & 15;       // A row / B column within tile
    const int half  = lane >> 4;       // 0: lanes 0-15, 1: lanes 16-31
    const int mtile0 = (blockIdx.x * 8 + wave) * 2;
    const int pBase  = mtile0 * 16;
    const int nBase  = blockIdx.y * 64;

    v8f acc[2][4];
#pragma unroll
    for (int a = 0; a < 2; ++a)
#pragma unroll
        for (int b = 0; b < 4; ++b) acc[a][b] = (v8f){0.f,0.f,0.f,0.f,0.f,0.f,0.f,0.f};

    const int off8  = half * 8;   // A lane K offset (elements)
    const int off16 = half * 8;   // B lane K offset (elements)

    for (int k = 0; k < 3; ++k) {
        const int l  = m & (LSPAN - 1);
        const int lp = l + k - 1;                       // SAME-padded neighbor
        const bool valid = (unsigned)lp < (unsigned)LSPAN;
        const int r0 = valid ? (pBase + m + (k - 1))      : padRow;
        const int r1 = valid ? (pBase + 16 + m + (k - 1)) : padRow;
        const unsigned short* aH0 = embHi + (size_t)r0 * kH;
        const unsigned short* aL0 = embLo + (size_t)r0 * kH;
        const unsigned short* aH1 = embHi + (size_t)r1 * kH;
        const unsigned short* aL1 = embLo + (size_t)r1 * kH;

        for (int chunk = 0; chunk < 4; ++chunk) {
            const int Kc = k * kH + chunk * KCH;        // global K base of this chunk

            __syncthreads();                            // previous chunk fully consumed
            // Fill B panel: 2 planes x 64 rows x 192 elems = 3072 x 16B, 12 per thread.
#pragma unroll
            for (int j = 0; j < 12; ++j) {
                int cidx  = tid + j * 256;              // [0, 3072)
                int plane = cidx / 1536;
                int r     = cidx - plane * 1536;        // [0, 1536): 64 rows * 24 chunks
                int ocl   = r / 24;
                int kc8   = (r - ocl * 24) * 8;         // [0, 192) step 8
                const unsigned short* srcP = plane ? wLo : wHi;
                const unsigned short* src  =
                    srcP + (size_t)(nBase + ocl) * kK3 + Kc + kc8;
                unsigned short* dst = &ldsB[plane * LPLN + ocl * LROW + kc8];
                cp16_async(src, dst);
            }
            cp_async_wait();
            __syncthreads();                            // panel visible to all waves

#pragma unroll
            for (int kk = 0; kk < 6; ++kk) {
                const int Kb = Kc + kk * 32;            // global K (A addressing)
                const int kb = kk * 32;                 // LDS-local K
                AFrag ah0, al0, ah1, al1;
                ah0.u[0] = *(const u32x4*)(aH0 + Kb + off8);
                ah0.u[1] = *(const u32x4*)(aH0 + Kb + off8 + 16);
                al0.u[0] = *(const u32x4*)(aL0 + Kb + off8);
                al0.u[1] = *(const u32x4*)(aL0 + Kb + off8 + 16);
                ah1.u[0] = *(const u32x4*)(aH1 + Kb + off8);
                ah1.u[1] = *(const u32x4*)(aH1 + Kb + off8 + 16);
                al1.u[0] = *(const u32x4*)(aL1 + Kb + off8);
                al1.u[1] = *(const u32x4*)(aL1 + Kb + off8 + 16);
#pragma unroll
                for (int nt = 0; nt < 4; ++nt) {
                    const unsigned short* lbH =
                        &ldsB[(nt * 16 + m) * LROW + kb + off16];
                    const unsigned short* lbL = lbH + LPLN;
                    AFrag bh, bl;
                    bh.u[0] = *(const u32x4*)(lbH);
                    bh.u[1] = *(const u32x4*)(lbH + 16);
                    bl.u[0] = *(const u32x4*)(lbL);
                    bl.u[1] = *(const u32x4*)(lbL + 16);
                    acc[0][nt] = wmma_bf16(ah0.v, bh.v, acc[0][nt]);
                    acc[0][nt] = wmma_bf16(ah0.v, bl.v, acc[0][nt]);
                    acc[0][nt] = wmma_bf16(al0.v, bh.v, acc[0][nt]);
                    acc[1][nt] = wmma_bf16(ah1.v, bh.v, acc[1][nt]);
                    acc[1][nt] = wmma_bf16(ah1.v, bl.v, acc[1][nt]);
                    acc[1][nt] = wmma_bf16(al1.v, bh.v, acc[1][nt]);
                }
            }
        }
    }

    // Epilogue: +bias, ReLU, masked mean over span length.
#pragma unroll
    for (int mt = 0; mt < 2; ++mt) {
        const int tile = mtile0 + mt;
        if (LSPAN == 16) {
            const int span = tile;                    // one span per 16-row tile
            const int len  = lens[span];
#pragma unroll
            for (int nt = 0; nt < 4; ++nt) {
                const int oc = nBase + nt * 16 + m;
                const float bias = convB[oc];
                float s = 0.f;
#pragma unroll
                for (int r = 0; r < 8; ++r) {
                    const int lpos = r + half * 8;    // C/D layout: M = r + 8*half
                    float y = acc[mt][nt][r] + bias;
                    y = y > 0.f ? y : 0.f;
                    if (lpos < len) s += y;
                }
                s += __shfl_xor(s, 16, 32);           // combine the two row halves
                if (lane < 16)
                    pooled[(size_t)span * kH + oc] = s / (float)len;
            }
        } else {                                       // LSPAN == 8: 2 spans per tile
            const int span = tile * 2 + half;          // lane halves own distinct spans
            const int len  = lens[span];
#pragma unroll
            for (int nt = 0; nt < 4; ++nt) {
                const int oc = nBase + nt * 16 + m;
                const float bias = convB[oc];
                float s = 0.f;
#pragma unroll
                for (int r = 0; r < 8; ++r) {          // l == r within this span
                    float y = acc[mt][nt][r] + bias;
                    y = y > 0.f ? y : 0.f;
                    if (r < len) s += y;
                }
                pooled[(size_t)span * kH + oc] = s / (float)len;
            }
        }
    }
}

// ---------------- kernel 4: 2-wide linear head + ReLU ----------------
__global__ void schema_head(const float* __restrict__ pooled,
                            const float* __restrict__ hw,
                            const float* __restrict__ hb,
                            float* __restrict__ out) {
    const int s    = blockIdx.x;
    const int lane = threadIdx.x;
    float s0 = 0.f, s1 = 0.f;
    const float* p = pooled + (size_t)s * kH;
    for (int c = lane; c < kH; c += 32) {
        float v = p[c];
        s0 += v * hw[c];
        s1 += v * hw[kH + c];
    }
#pragma unroll
    for (int off = 16; off > 0; off >>= 1) {
        s0 += __shfl_xor(s0, off, 32);
        s1 += __shfl_xor(s1, off, 32);
    }
    if (lane == 0) {
        float a = s0 + hb[0];
        float b = s1 + hb[1];
        out[(size_t)s * 2 + 0] = a > 0.f ? a : 0.f;
        out[(size_t)s * 2 + 1] = b > 0.f ? b : 0.f;
    }
}

// ---------------- launch ----------------
extern "C" void kernel_launch(void* const* d_in, const int* in_sizes, int n_in,
                              void* d_out, int out_size, void* d_ws, size_t ws_size,
                              hipStream_t stream) {
    const float* hidden      = (const float*)d_in[0];
    const int*   table_ids   = (const int*)d_in[1];
    const int*   table_len   = (const int*)d_in[2];
    const int*   col_ids     = (const int*)d_in[3];
    const int*   col_len     = (const int*)d_in[4];
    const float* t_conv_w    = (const float*)d_in[5];
    const float* t_conv_b    = (const float*)d_in[6];
    const float* t_head_w    = (const float*)d_in[7];
    const float* t_head_b    = (const float*)d_in[8];
    const float* c_conv_w    = (const float*)d_in[9];
    const float* c_conv_b    = (const float*)d_in[10];
    const float* c_head_w    = (const float*)d_in[11];
    const float* c_head_b    = (const float*)d_in[12];
    float* out = (float*)d_out;

    char* ws = (char*)d_ws;
    unsigned short* embT_hi = (unsigned short*)(ws + OFF_EMBT_HI);
    unsigned short* embT_lo = (unsigned short*)(ws + OFF_EMBT_LO);
    unsigned short* embC_hi = (unsigned short*)(ws + OFF_EMBC_HI);
    unsigned short* embC_lo = (unsigned short*)(ws + OFF_EMBC_LO);
    unsigned short* wT_hi   = (unsigned short*)(ws + OFF_WT_HI);
    unsigned short* wT_lo   = (unsigned short*)(ws + OFF_WT_LO);
    unsigned short* wC_hi   = (unsigned short*)(ws + OFF_WC_HI);
    unsigned short* wC_lo   = (unsigned short*)(ws + OFF_WC_LO);
    float* poolT = (float*)(ws + OFF_POOL_T);
    float* poolC = (float*)(ws + OFF_POOL_C);

    // 1) gather + split (includes zero pad row at index P_*)
    {
        int totT = (P_T + 1) * (kH / 4);
        int totC = (P_C + 1) * (kH / 4);
        schema_gather_split<<<(totT + 255) / 256, 256, 0, stream>>>(
            hidden, table_ids, table_len, embT_hi, embT_lo, kT, kLT, P_T);
        schema_gather_split<<<(totC + 255) / 256, 256, 0, stream>>>(
            hidden, col_ids, col_len, embC_hi, embC_lo, kC, kLC, P_C);
    }
    // 2) weight reorder + split
    {
        int tot = kH * kK3;                   // 1,769,472 = 6912 * 256 exactly
        schema_prep_w<<<tot / 256, 256, 0, stream>>>(t_conv_w, wT_hi, wT_lo);
        schema_prep_w<<<tot / 256, 256, 0, stream>>>(c_conv_w, wC_hi, wC_lo);
    }
    // 3) WMMA conv-GEMM + pool (256 positions x 64 channels per block)
    schema_conv_gemm_pool<kLT><<<dim3(P_T / 256, kH / 64), 256, 0, stream>>>(
        embT_hi, embT_lo, wT_hi, wT_lo, t_conv_b, table_len, poolT, P_T);
    schema_conv_gemm_pool<kLC><<<dim3(P_C / 256, kH / 64), 256, 0, stream>>>(
        embC_hi, embC_lo, wC_hi, wC_lo, c_conv_b, col_len, poolC, P_C);
    // 4) heads (table logits first, then column logits, flat)
    schema_head<<<NS_T, 32, 0, stream>>>(poolT, t_head_w, t_head_b, out);
    schema_head<<<NS_C, 32, 0, stream>>>(poolC, c_head_w, c_head_b, out + (size_t)NS_T * 2);

    (void)in_sizes; (void)n_in; (void)out_size; (void)ws_size;
}
